// DeformableBlock_3556232921201
// MI455X (gfx1250) — matrix operand order repaired and checked by reading, not compiled, not run on hardware
//
#include <hip/hip_runtime.h>

// ---------------- problem constants ----------------
#define BATCH  4
#define CH     256          // CIN == COUT
#define HIMG   64
#define WIMG   64
#define HW     4096         // HIMG*WIMG
#define NTAP   9            // 3x3 kernel taps
#define OFFC   18           // 2*NTAP offset channels
#define KTOT   (NTAP*CH)    // 2304 contraction length
#define GROUPS 32
#define GEPS   1e-5f
#define TILE_M 32           // pixels per workgroup

typedef __attribute__((ext_vector_type(16))) __bf16 v16bf;
typedef __attribute__((ext_vector_type(8)))  float  v8f;

// ------------- bf16 <-> f32 helpers (bit-level, RNE) -------------
__device__ __forceinline__ unsigned short f2bf(float f) {
    unsigned u = __builtin_bit_cast(unsigned, f);
    u += 0x7fffu + ((u >> 16) & 1u);
    return (unsigned short)(u >> 16);
}
__device__ __forceinline__ float bf2f(unsigned short h) {
    unsigned u = ((unsigned)h) << 16;
    return __builtin_bit_cast(float, u);
}

// ---------------- kernel 0: zero GN stats ----------------
__global__ __launch_bounds__(256) void init_stats_kernel(float* stats) {
    stats[threadIdx.x] = 0.0f;   // 256 floats: [0..127]=sum, [128..255]=sumsq
}

// ---------------- kernel 1: x [B][C][H][W] f32 -> x_t [B][HW][C] bf16 ----------------
__global__ __launch_bounds__(256) void transpose_x_kernel(const float* __restrict__ x,
                                                          unsigned short* __restrict__ xt) {
    int idx = blockIdx.x * 256 + threadIdx.x;      // over B*HW*CH = 4,194,304
    int c   = idx & (CH - 1);
    int p   = (idx >> 8) & (HW - 1);
    int b   = idx >> 20;
    xt[idx] = f2bf(x[((size_t)(b * CH + c)) * HW + p]);
}

// ---------------- kernel 2: pack deform_w into WMMA B-fragment-major bf16 ----------------
// w_frag layout: [tap(9)][kchunk(8)][ntile(16)][lane(32)][elem(16)]
// element (lane l, elem e) = B[K = 16*(l/16)+e][N = l%16] of the 32x16 B tile
// global row kk = tap*256 + kchunk*32 + K  (i.e. c = kchunk*32 + K), col o = ntile*16 + N
__global__ __launch_bounds__(256) void pack_wfrag_kernel(const float* __restrict__ dw,
                                                         unsigned short* __restrict__ wf) {
    int i  = blockIdx.x * 256 + threadIdx.x;       // over 9*8*16*32*16 = 589,824
    int e  = i & 15;
    int l  = (i >> 4) & 31;
    int nt = (i >> 9) & 15;
    int kc = (i >> 13) & 7;
    int t  = i >> 16;
    int K  = 16 * (l >> 4) + e;
    int c  = kc * 32 + K;
    int o  = nt * 16 + (l & 15);
    wf[i] = f2bf(dw[((size_t)(o * CH + c)) * NTAP + t]);
}

// ---------------- kernel 3: offset conv3x3 (256 -> 18) + bias ----------------
__global__ __launch_bounds__(256) void offset_conv_kernel(const float* __restrict__ x,
                                                          const float* __restrict__ ow,
                                                          const float* __restrict__ ob,
                                                          float* __restrict__ off) {
    __shared__ float sw[32 * NTAP * OFFC];         // 20.25 KB weight tile for 32 channels
    int wg  = blockIdx.x;                          // 64 blocks: b*16 + pixtile
    int b   = wg >> 4;
    int pix = ((wg & 15) << 8) + threadIdx.x;      // 256 pixels per block
    int py  = pix >> 6, px = pix & 63;

    float acc[OFFC];
#pragma unroll
    for (int oc = 0; oc < OFFC; ++oc) acc[oc] = 0.0f;

    for (int c0 = 0; c0 < CH; c0 += 32) {
        __syncthreads();
        for (int i = threadIdx.x; i < 32 * NTAP * OFFC; i += 256) {
            int c  = i / (NTAP * OFFC);
            int r  = i % (NTAP * OFFC);
            int t  = r / OFFC;
            int oc = r % OFFC;
            sw[i] = ow[((size_t)(oc * CH + c0 + c)) * NTAP + t];
        }
        __syncthreads();
        for (int c = 0; c < 32; ++c) {
            const float* xb = x + ((size_t)(b * CH + c0 + c)) * HW;
#pragma unroll
            for (int t = 0; t < NTAP; ++t) {
                int yy = py - 1 + t / 3;
                int xx = px - 1 + t % 3;
                float xv = (yy >= 0 && yy < HIMG && xx >= 0 && xx < WIMG)
                               ? xb[yy * WIMG + xx] : 0.0f;
                const float* wrow = &sw[(c * NTAP + t) * OFFC];
#pragma unroll
                for (int oc = 0; oc < OFFC; ++oc) acc[oc] += xv * wrow[oc];
            }
        }
    }
#pragma unroll
    for (int oc = 0; oc < OFFC; ++oc)
        off[((size_t)(b * OFFC + oc)) * HW + pix] = acc[oc] + ob[oc];
}

// ---------------- kernel 4: fused bilinear sampling + WMMA GEMM ----------------
// Workgroup = 256 threads (8 waves, wave32). Tile: 32 pixels x 256 out-channels.
// Per tap: build A tile (32 x 256 bf16) in LDS, then each wave runs 2 M-tiles x
// 2 N-tiles over 8 K-chunks of v_wmma_f32_16x16x32_bf16 (each B fragment is
// reused for 2 WMMAs -> halves L2 weight traffic vs a 16-pixel tile).
__global__ __launch_bounds__(256) void deform_wmma_kernel(const unsigned short* __restrict__ xt,
                                                          const unsigned short* __restrict__ wf,
                                                          const float* __restrict__ off,
                                                          float* __restrict__ y,
                                                          float* __restrict__ gsum,
                                                          float* __restrict__ gsq) {
    __shared__ int            s_idx[NTAP][TILE_M][4];
    __shared__ float          s_w[NTAP][TILE_M][4];
    __shared__ unsigned short s_a[TILE_M][CH + 8];     // +8 pad vs bank conflicts

    const int wg   = blockIdx.x;                       // 512 = B * (HW/32)
    const int b    = wg >> 7;
    const int pix0 = (wg & 127) << 5;
    const int tid  = threadIdx.x;
    const int lane = tid & 31;
    const int wave = tid >> 5;
    const int half = lane >> 4;
    const int mrow = lane & 15;

    // ---- phase 1: bilinear sampling parameters for 32 pixels x 9 taps ----
    for (int i = tid; i < NTAP * TILE_M; i += 256) {
        int tap = i >> 5, m = i & (TILE_M - 1);
        int pix = pix0 + m;
        int py_i = pix >> 6, px_i = pix & 63;
        float offy = off[((size_t)(b * OFFC + 2 * tap)) * HW + pix];
        float offx = off[((size_t)(b * OFFC + 2 * tap + 1)) * HW + pix];
        float py = (float)(py_i - 1 + tap / 3) + offy;
        float px = (float)(px_i - 1 + tap % 3) + offx;
        float fy = floorf(py), fx = floorf(px);
        int y0 = (int)fy, x0 = (int)fx;
        float ty = py - fy, tx = px - fx;
        float cw[4] = { (1.f - ty) * (1.f - tx), (1.f - ty) * tx,
                        ty * (1.f - tx),         ty * tx };
        int cy[4] = { y0, y0, y0 + 1, y0 + 1 };
        int cx[4] = { x0, x0 + 1, x0, x0 + 1 };
#pragma unroll
        for (int j = 0; j < 4; ++j) {
            bool v = (cy[j] >= 0) & (cy[j] < HIMG) & (cx[j] >= 0) & (cx[j] < WIMG);
            s_idx[tap][m][j] = v ? ((cy[j] * WIMG + cx[j]) * CH) : 0;
            s_w[tap][m][j]   = v ? cw[j] : 0.0f;
        }
    }
    __syncthreads();

    v8f acc[2][2] = {};                                 // [m-tile][n-tile]
    const unsigned short* xtb = xt + (size_t)b * HW * CH;

    for (int tap = 0; tap < NTAP; ++tap) {
        __syncthreads();                                // previous A tile fully consumed
        // ---- cooperative sampled-A tile build: 32 pixels x 256 channels ----
        for (int j = tid; j < TILE_M * CH; j += 256) {
            int m = j >> 8, c = j & 255;
            const int*   ip = s_idx[tap][m];
            const float* wp = s_w[tap][m];
            float v = wp[0] * bf2f(xtb[ip[0] + c]) + wp[1] * bf2f(xtb[ip[1] + c])
                    + wp[2] * bf2f(xtb[ip[2] + c]) + wp[3] * bf2f(xtb[ip[3] + c]);
            s_a[m][c] = f2bf(v);
        }
        __syncthreads();

        const unsigned short* wtap = wf + (size_t)tap * (8 * 16 * 32 * 16);
        if (tap + 1 < NTAP)                              // global_prefetch_b8 next tap's frags
            __builtin_prefetch(wtap + (size_t)(8 * 16 * 32 * 16) + (size_t)wave * 1024, 0, 1);

#pragma unroll
        for (int kc = 0; kc < 8; ++kc) {
            // ---- A fragments (2 M-tiles), contiguous k order k = 16*half + e (matches B) ----
            union { v16bf v; unsigned short u[16]; } A0, A1;
#pragma unroll
            for (int vv = 0; vv < 8; ++vv) {
                int k0 = 2 * vv + 16 * half;
                A0.u[2 * vv]     = s_a[mrow][kc * 32 + k0];
                A0.u[2 * vv + 1] = s_a[mrow][kc * 32 + k0 + 1];
                A1.u[2 * vv]     = s_a[16 + mrow][kc * 32 + k0];
                A1.u[2 * vv + 1] = s_a[16 + mrow][kc * 32 + k0 + 1];
            }
            // ---- B fragments: contiguous 32B per lane from pre-swizzled weights ----
            const unsigned short* bbase = wtap + (size_t)kc * (16 * 32 * 16);
            union { v16bf v; uint4 q[2]; } B0, B1;
            const uint4* p0 = (const uint4*)(bbase + ((size_t)(2 * wave)     * 32 + lane) * 16);
            const uint4* p1 = (const uint4*)(bbase + ((size_t)(2 * wave + 1) * 32 + lane) * 16);
            B0.q[0] = p0[0]; B0.q[1] = p0[1];
            B1.q[0] = p1[0]; B1.q[1] = p1[1];

            acc[0][0] = __builtin_amdgcn_wmma_f32_16x16x32_bf16(false, A0.v, false, B0.v,
                                                                (short)0, acc[0][0], false, false);
            acc[0][1] = __builtin_amdgcn_wmma_f32_16x16x32_bf16(false, A0.v, false, B1.v,
                                                                (short)0, acc[0][1], false, false);
            acc[1][0] = __builtin_amdgcn_wmma_f32_16x16x32_bf16(false, A1.v, false, B0.v,
                                                                (short)0, acc[1][0], false, false);
            acc[1][1] = __builtin_amdgcn_wmma_f32_16x16x32_bf16(false, A1.v, false, B1.v,
                                                                (short)0, acc[1][1], false, false);
        }
    }

    // ---- epilogue: store NCHW f32 (32B contiguous per lane) + GN partial sums ----
#pragma unroll
    for (int mt = 0; mt < 2; ++mt) {
#pragma unroll
        for (int nt = 0; nt < 2; ++nt) {
            v8f a = acc[mt][nt];
            int o = (2 * wave + nt) * 16 + mrow;        // column (out channel) of C/D tile
            float* dst = y + ((size_t)(b * CH + o)) * HW + pix0 + mt * 16 + 8 * half;
            float4 lo = make_float4(a[0], a[1], a[2], a[3]);
            float4 hi = make_float4(a[4], a[5], a[6], a[7]);
            ((float4*)dst)[0] = lo;
            ((float4*)dst)[1] = hi;
            float s  = a[0] + a[1] + a[2] + a[3] + a[4] + a[5] + a[6] + a[7];
            float sq = a[0]*a[0] + a[1]*a[1] + a[2]*a[2] + a[3]*a[3]
                     + a[4]*a[4] + a[5]*a[5] + a[6]*a[6] + a[7]*a[7];
            int g = b * GROUPS + (o >> 3);              // 8 channels per group
            atomicAdd(&gsum[g], s);
            atomicAdd(&gsq[g], sq);
        }
    }
}

// ---------------- kernel 5: GroupNorm + ReLU in place on d_out ----------------
__global__ __launch_bounds__(256) void gn_relu_kernel(float* __restrict__ y,
                                                      const float* __restrict__ stats,
                                                      const float* __restrict__ gamma,
                                                      const float* __restrict__ beta) {
    int i4   = blockIdx.x * 256 + threadIdx.x;      // over (B*CH*HW)/4 float4s
    int base = i4 << 2;
    int o    = (base >> 12) & 255;
    int b    = base >> 20;
    int g    = b * GROUPS + (o >> 3);
    const float invN = 1.0f / (float)(8 * HW);
    float mu  = stats[g] * invN;
    float var = stats[128 + g] * invN - mu * mu;
    float sc  = rsqrtf(var + GEPS) * gamma[o];
    float sh  = beta[o] - mu * sc;
    float4 v = ((float4*)y)[i4];
    v.x = fmaxf(v.x * sc + sh, 0.0f);
    v.y = fmaxf(v.y * sc + sh, 0.0f);
    v.z = fmaxf(v.z * sc + sh, 0.0f);
    v.w = fmaxf(v.w * sc + sh, 0.0f);
    ((float4*)y)[i4] = v;
}

// ---------------- host launcher ----------------
extern "C" void kernel_launch(void* const* d_in, const int* in_sizes, int n_in,
                              void* d_out, int out_size, void* d_ws, size_t ws_size,
                              hipStream_t stream) {
    const float* x  = (const float*)d_in[0];   // [4,256,64,64]
    const float* ow = (const float*)d_in[1];   // [18,256,3,3]
    const float* ob = (const float*)d_in[2];   // [18]
    const float* dw = (const float*)d_in[3];   // [256,256,3,3]
    const float* gg = (const float*)d_in[4];   // [256]
    const float* gb = (const float*)d_in[5];   // [256]
    float* out = (float*)d_out;                // [4,256,64,64] f32

    char* ws = (char*)d_ws;
    size_t XT_OFF   = 0;                                          // bf16 x_t: 8 MB
    size_t WF_OFF   = XT_OFF + (size_t)BATCH * HW * CH * 2;       // bf16 w_frag: 1.125 MB
    size_t OFF_OFF  = WF_OFF + (size_t)KTOT * CH * 2;             // f32 offsets: 1.125 MB
    size_t STAT_OFF = OFF_OFF + (size_t)BATCH * OFFC * HW * 4;    // 256 f32 stats

    unsigned short* xt    = (unsigned short*)(ws + XT_OFF);
    unsigned short* wfrag = (unsigned short*)(ws + WF_OFF);
    float*          offs  = (float*)(ws + OFF_OFF);
    float*          stats = (float*)(ws + STAT_OFF);

    init_stats_kernel<<<1, 256, 0, stream>>>(stats);
    transpose_x_kernel<<<(BATCH * HW * CH) / 256, 256, 0, stream>>>(x, xt);
    pack_wfrag_kernel<<<(NTAP * 8 * 16 * 32 * 16) / 256, 256, 0, stream>>>(dw, wfrag);
    offset_conv_kernel<<<BATCH * (HW / 256), 256, 0, stream>>>(x, ow, ob, offs);
    deform_wmma_kernel<<<BATCH * (HW / TILE_M), 256, 0, stream>>>(xt, wfrag, offs, out,
                                                                  stats, stats + 128);
    gn_relu_kernel<<<(BATCH * CH * HW) / 4 / 256, 256, 0, stream>>>(out, stats, gg, gb);
}
